// Attention_43233140801716
// MI455X (gfx1250) — compile-verified, matrix operand order checked by reading, not run on hardware
//
#include <hip/hip_runtime.h>
#include <stdint.h>

#define BB 4
#define NN 4096
#define DD 256

typedef __attribute__((ext_vector_type(16))) __bf16          v16bf;
typedef __attribute__((ext_vector_type(4)))  __bf16          v4bf;
typedef __attribute__((ext_vector_type(8)))  float           v8f;
typedef __attribute__((ext_vector_type(16))) unsigned short  u16x16;

static __device__ __forceinline__ unsigned short bf(float f) {
  __bf16 h = (__bf16)f;                    // native v_cvt_pk_bf16_f32
  return __builtin_bit_cast(unsigned short, h);
}

static __device__ __forceinline__ v16bf ld_bf16x16(const unsigned short* p) {
  return __builtin_bit_cast(v16bf, *(const u16x16*)p);
}

// ---------------------------------------------------------------------------
// Kernel 0: bulk f32 -> bf16 conversion (x and the three weight matrices).
// ---------------------------------------------------------------------------
__global__ __launch_bounds__(256) void cvt_kernel(
    const float* __restrict__ src, unsigned short* __restrict__ dst, int nvec)
{
  int i = blockIdx.x * 256 + threadIdx.x;
  if (i < nvec) {
    float4 f = ((const float4*)src)[i];
    v4bf h;
    h.x = (__bf16)f.x; h.y = (__bf16)f.y;
    h.z = (__bf16)f.z; h.w = (__bf16)f.w;
    *(v4bf*)(dst + 4 * (size_t)i) = h;
  }
}

// ---------------------------------------------------------------------------
// Kernel 1: fused QKV projection from bf16 x and bf16 W.  No LDS needed:
// each wave's A rows are private (lane lr / lr+16 read disjoint 32B halves).
// q,k row-major [b*n][d]; v written transposed [b][d][n].
// Destination factored as obase + col*colStride + i*rowStride, resolved once
// -> fully branch-free main loop.  B tiles preloaded into bb[8] so the load
// clause overlaps the WMMA chain.
// ---------------------------------------------------------------------------
__global__ __launch_bounds__(256, 1) void qkv_kernel(
    const unsigned short* __restrict__ xbf,
    const unsigned short* __restrict__ wqb, const float* __restrict__ bq,
    const unsigned short* __restrict__ wkb, const float* __restrict__ bk,
    const unsigned short* __restrict__ wvb, const float* __restrict__ bv,
    unsigned short* __restrict__ qw, unsigned short* __restrict__ kw,
    unsigned short* __restrict__ vtw)
{
  const int sel = blockIdx.y;
  const unsigned short* W = sel == 0 ? wqb : sel == 1 ? wkb : wvb;
  const float* bias       = sel == 0 ? bq  : sel == 1 ? bk  : bv;

  const int tid  = threadIdx.x;
  const int wave = tid >> 5, lane = tid & 31;
  const int half = lane >> 4, lr = lane & 15;
  const size_t row0 = (size_t)blockIdx.x * 128;

  // this lane's first output row (C layout: VGPR i -> row_base + i)
  const size_t row_base = row0 + wave * 16 + half * 8;
  const size_t bidx     = row_base >> 12;          // whole block in one batch
  const size_t n_base   = row_base & (NN - 1);

  // destination addressing resolved ONCE: addr = obase + col*colS + i*rowS
  unsigned short* obase;
  size_t colS, rowS;
  if (sel == 0)      { obase = qw + row_base * DD;                colS = 1;  rowS = DD; }
  else if (sel == 1) { obase = kw + row_base * DD;                colS = 1;  rowS = DD; }
  else               { obase = vtw + bidx * DD * NN + n_base;     colS = NN; rowS = 1;  }

  // A operand: token rows, K=256 in 8 chunks of 32
  v16bf a[8];
  {
    const unsigned short* xp = xbf + (row0 + wave * 16 + lr) * DD + half * 16;
    #pragma unroll
    for (int kk = 0; kk < 8; ++kk) a[kk] = ld_bf16x16(xp + kk * 32);
  }

  #pragma unroll 1
  for (int c = 0; c < 16; ++c) {          // 16 output-column tiles
    const unsigned short* wp = W + (size_t)(c * 16 + lr) * DD + half * 16;

    // preload all 8 B tiles -> load clause overlaps the WMMA chain
    v16bf bb[8];
    #pragma unroll
    for (int kk = 0; kk < 8; ++kk) bb[kk] = ld_bf16x16(wp + kk * 32);

    v8f acc = {};
    #pragma unroll
    for (int kk = 0; kk < 8; ++kk)
      acc = __builtin_amdgcn_wmma_f32_16x16x32_bf16(false, a[kk], false, bb[kk],
                                                    (short)0, acc, false, false);

    const int col = c * 16 + lr;
    const float bval = bias[col];
    unsigned short* dst = obase + (size_t)col * colS;
    #pragma unroll
    for (int i = 0; i < 8; ++i)
      dst[(size_t)i * rowS] = bf(acc[i] + bval);
  }
}

// ---------------------------------------------------------------------------
// Kernel 2: flash attention, double-buffered async K/V staging.
// Block = 4 waves = 64 queries of one batch; 32-key chunks.
// S^T = K x Q^T (queries in lanes -> cheap softmax, P already A-oriented).
// ---------------------------------------------------------------------------
__global__ __launch_bounds__(128, 1) void attn_kernel(
    const unsigned short* __restrict__ qw,
    const unsigned short* __restrict__ kw,
    const unsigned short* __restrict__ vtw,
    const float* __restrict__ scale,
    float* __restrict__ out)
{
  __shared__ __align__(32) unsigned short ks[2][32 * DD];   // [key][d]  2x16KB
  __shared__ __align__(32) unsigned short vs[2][DD * 32];   // [d][key]  2x16KB

  const int b   = blockIdx.y;
  const int q0  = blockIdx.x * 64;
  const int tid = threadIdx.x;
  const int wave = tid >> 5, lane = tid & 31;
  const int half = lane >> 4, lr = lane & 15;
  const float inv_scale = 1.0f / scale[0];

  // Q in registers: lane lr <-> query lr (B-operand orientation)
  v16bf qreg[8];
  {
    const unsigned short* qp =
        qw + (size_t)(b * NN + q0 + wave * 16 + lr) * DD + half * 16;
    #pragma unroll
    for (int kk = 0; kk < 8; ++kk) qreg[kk] = ld_bf16x16(qp + kk * 32);
  }

  v8f o[16];
  #pragma unroll
  for (int c = 0; c < 16; ++c) o[c] = (v8f){};
  float m_run = -3.0e38f, l_run = 0.0f;

  const unsigned short* kg = kw  + (size_t)b * NN * DD;
  const unsigned long long vsrc =
      (unsigned long long)(uintptr_t)(vtw + (size_t)b * DD * NN);
  const unsigned lds_k0 = (unsigned)(size_t)&ks[0][0];
  const unsigned lds_v0 = (unsigned)(size_t)&vs[0][0];

  auto stage = [&](int kc, int buf) {
    const unsigned long long ksrc =
        (unsigned long long)(uintptr_t)(kg + (size_t)(kc * 32) * DD);
    const unsigned lk = lds_k0 + (unsigned)buf * (32 * DD * 2);
    #pragma unroll
    for (int i = 0; i < 8; ++i) {
      unsigned seg = tid + i * 128;                 // 1024 x 16B = 16 KB
      asm volatile("global_load_async_to_lds_b128 %0, %1, %2"
                   :: "v"(lk + seg * 16), "v"(seg * 16), "s"(ksrc) : "memory");
    }
    const unsigned lv = lds_v0 + (unsigned)buf * (DD * 32 * 2);
    #pragma unroll
    for (int i = 0; i < 8; ++i) {
      unsigned idx = tid + i * 128;                 // 256 d-rows x 4 segs
      unsigned d = idx >> 2, s = idx & 3;
      asm volatile("global_load_async_to_lds_b128 %0, %1, %2"
                   :: "v"(lv + d * 64 + s * 16),
                      "v"((d * NN + kc * 32) * 2 + s * 16), "s"(vsrc) : "memory");
    }
  };

  const int NC = NN / 32;
  stage(0, 0);

  for (int kc = 0; kc < NC; ++kc) {
    const int buf = kc & 1;
    if (kc + 1 < NC) {
      stage(kc + 1, buf ^ 1);                       // prefetch next chunk
      asm volatile("s_wait_asynccnt 0x10" ::: "memory");  // current 16 done
    } else {
      asm volatile("s_wait_asynccnt 0x0" ::: "memory");
    }
    __syncthreads();

    // ---- S^T = K_chunk x Q^T : two 16(key) x 16(query) f32 tiles ----
    v8f s0 = (v8f){}, s1 = (v8f){};
    #pragma unroll
    for (int kk = 0; kk < 8; ++kk) {
      v16bf a0 = ld_bf16x16(&ks[buf][(lr)      * DD + kk * 32 + half * 16]);
      v16bf a1 = ld_bf16x16(&ks[buf][(16 + lr) * DD + kk * 32 + half * 16]);
      s0 = __builtin_amdgcn_wmma_f32_16x16x32_bf16(false, a0, false, qreg[kk],
                                                   (short)0, s0, false, false);
      s1 = __builtin_amdgcn_wmma_f32_16x16x32_bf16(false, a1, false, qreg[kk],
                                                   (short)0, s1, false, false);
    }

    // ---- online softmax: lane lr = query; halves hold disjoint keys ----
    float mloc = -3.0e38f;
    #pragma unroll
    for (int i = 0; i < 8; ++i) {
      s0[i] *= inv_scale; s1[i] *= inv_scale;
      mloc = fmaxf(mloc, fmaxf(s0[i], s1[i]));
    }
    mloc = fmaxf(mloc, __shfl_xor(mloc, 16, 32));
    const float mnew  = fmaxf(m_run, mloc);
    const float alpha = __expf(m_run - mnew);
    float p0[8], p1[8], lsum = 0.0f;
    #pragma unroll
    for (int i = 0; i < 8; ++i) {
      p0[i] = __expf(s0[i] - mnew);
      p1[i] = __expf(s1[i] - mnew);
      lsum += p0[i] + p1[i];
    }
    lsum += __shfl_xor(lsum, 16, 32);
    l_run = l_run * alpha + lsum;
    m_run = mnew;

    // rescale O accumulators by per-query alpha (broadcast via shfl)
    #pragma unroll
    for (int i = 0; i < 8; ++i) {
      float ai = __shfl(alpha, i + half * 8, 32);
      #pragma unroll
      for (int c = 0; c < 16; ++c) o[c][i] *= ai;
    }

    // pack P (16 queries x 32 keys) into bf16 A-operand
    u16x16 pu;
    #pragma unroll
    for (int j = 0; j < 8; ++j) {
      float x0 = __shfl_xor(p0[j], 16, 32);
      float x1 = __shfl_xor(p1[j], 16, 32);
      pu[j]     = bf(half ? x1 : p0[j]);   // keys  0..15 (lo-half lanes)
      pu[j + 8] = bf(half ? p1[j] : x0);   // keys 16..31 (hi-half lanes)
    }
    v16bf pa = __builtin_bit_cast(v16bf, pu);

    // ---- O += P x V_chunk : B operand contiguous from d-major V^T ----
    #pragma unroll
    for (int c = 0; c < 16; ++c) {
      v16bf bv = ld_bf16x16(&vs[buf][(c * 16 + lr) * 32 + half * 16]);
      o[c] = __builtin_amdgcn_wmma_f32_16x16x32_bf16(false, pa, false, bv,
                                                     (short)0, o[c], false, false);
    }
    __syncthreads();   // all readers done before this buffer is restaged
  }

  // ---- epilogue: normalize by 1/l and store f32 ----
  const float linv = 1.0f / l_run;
  #pragma unroll
  for (int c = 0; c < 16; ++c) {
    #pragma unroll
    for (int i = 0; i < 8; ++i) {
      float li = __shfl(linv, i + half * 8, 32);
      size_t r = (size_t)(b * NN + q0 + wave * 16 + i + half * 8);
      out[r * DD + c * 16 + lr] = o[c][i] * li;
    }
  }
}

// ---------------------------------------------------------------------------
extern "C" void kernel_launch(void* const* d_in, const int* in_sizes, int n_in,
                              void* d_out, int out_size, void* d_ws, size_t ws_size,
                              hipStream_t stream)
{
  const float* x  = (const float*)d_in[0];
  const float* Wq = (const float*)d_in[1];
  const float* bq = (const float*)d_in[2];
  const float* Wk = (const float*)d_in[3];
  const float* bk = (const float*)d_in[4];
  const float* Wv = (const float*)d_in[5];
  const float* bv = (const float*)d_in[6];
  const float* sc = (const float*)d_in[7];

  const size_t BND = (size_t)BB * NN * DD;       // 4,194,304 elements
  unsigned short* qw  = (unsigned short*)d_ws;   // bf16 [B*N][D]
  unsigned short* kw  = qw  + BND;               // bf16 [B*N][D]
  unsigned short* vtw = kw  + BND;               // bf16 [B][D][N]
  unsigned short* xbf = vtw + BND;               // bf16 [B*N][D]
  unsigned short* wqb = xbf + BND;               // bf16 [D][D]
  unsigned short* wkb = wqb + (size_t)DD * DD;
  unsigned short* wvb = wkb + (size_t)DD * DD;

  // pre-pass: f32 -> bf16
  cvt_kernel<<<(int)(BND / 4 / 256), 256, 0, stream>>>(x, xbf, (int)(BND / 4));
  const int wvec = DD * DD / 4;                  // 16384
  cvt_kernel<<<(wvec + 255) / 256, 256, 0, stream>>>(Wq, wqb, wvec);
  cvt_kernel<<<(wvec + 255) / 256, 256, 0, stream>>>(Wk, wkb, wvec);
  cvt_kernel<<<(wvec + 255) / 256, 256, 0, stream>>>(Wv, wvb, wvec);

  qkv_kernel<<<dim3((BB * NN) / 128, 3), 256, 0, stream>>>(
      xbf, wqb, bq, wkb, bk, wvb, bv, qw, kw, vtw);
  attn_kernel<<<dim3(NN / 64, BB), 128, 0, stream>>>(
      qw, kw, vtw, sc, (float*)d_out);
}
